// ClaimEncoder_58849641890598
// MI455X (gfx1250) — matrix operand order repaired and checked by reading, not compiled, not run on hardware
//
#include <hip/hip_runtime.h>

typedef __attribute__((ext_vector_type(16))) __bf16 v16bf;
typedef __attribute__((ext_vector_type(8)))  __bf16 v8bf;
typedef __attribute__((ext_vector_type(8)))  float  v8f;

#define B_ROWS   131072
#define FEATN    896      // 2*256 + 3*128
#define CEDN     512
#define MT       32       // rows per block
#define FSTRIDE  904      // padded feature row stride (bf16): 452 dwords % 64 banks = 4
#define HSTRIDE  520      // padded h row stride: 260 dwords % 64 = 4

struct Params {
  const float* in[27];   // d_in[0..26]: batch inputs + encoder param vectors
  const float* b1;
  const float* b2;
  const __bf16* W1T;     // [CEDN][FEATN]  (W1 transposed, bf16)
  const __bf16* W2T;     // [CEDN][CEDN]
  float* out;
};

// ---------------------------------------------------------------------------
// Prep: W1 (896x512) -> W1T bf16 (512x896), W2 (512x512) -> W2T bf16 (512x512)
// ---------------------------------------------------------------------------
__global__ void prep_weights(const float* __restrict__ W1, const float* __restrict__ W2,
                             __bf16* __restrict__ W1T, __bf16* __restrict__ W2T) {
  const int total1 = CEDN * FEATN;
  const int total2 = CEDN * CEDN;
  for (int i = blockIdx.x * blockDim.x + threadIdx.x; i < total1 + total2;
       i += gridDim.x * blockDim.x) {
    if (i < total1) {
      int n = i / FEATN, k = i - n * FEATN;
      W1T[i] = (__bf16)W1[k * CEDN + n];
    } else {
      int j = i - total1;
      int n = j / CEDN, k = j - n * CEDN;
      W2T[j] = (__bf16)W2[k * CEDN + n];
    }
  }
}

union Frag { v16bf v; v8bf p[2]; };

// One GEMM stage: (32 x K) bf16 tile in LDS  x  K x 64 slice of W^T (global/L2)
// Each wave owns 2 M-tiles x 4 N-tiles. B fragments are double-buffered in
// registers so L2 latency overlaps the WMMA chain.
template<int K>
__device__ __forceinline__ void gemm_stage(const __bf16* __restrict__ a0p,
                                           const __bf16* __restrict__ a1p,
                                           const __bf16* __restrict__ wp,
                                           v8f (&acc)[2][4]) {
  Frag bcur[4], bnxt[4], a0, a1;
#pragma unroll
  for (int t = 0; t < 4; ++t) {
    const __bf16* w = wp + (size_t)t * 16 * K;
    bcur[t].p[0] = *(const v8bf*)(w);
    bcur[t].p[1] = *(const v8bf*)(w + 16);
  }
  int kk = 0;
  for (; kk < K - 32; kk += 32) {
    a0.p[0] = *(const v8bf*)(a0p + kk);
    a0.p[1] = *(const v8bf*)(a0p + kk + 16);
    a1.p[0] = *(const v8bf*)(a1p + kk);
    a1.p[1] = *(const v8bf*)(a1p + kk + 16);
#pragma unroll
    for (int t = 0; t < 4; ++t) {                 // prefetch next k-step
      const __bf16* w = wp + (size_t)t * 16 * K + kk + 32;
      bnxt[t].p[0] = *(const v8bf*)(w);
      bnxt[t].p[1] = *(const v8bf*)(w + 16);
    }
#pragma unroll
    for (int t = 0; t < 4; ++t) {
      acc[0][t] = __builtin_amdgcn_wmma_f32_16x16x32_bf16(
          false, a0.v, false, bcur[t].v, (short)0, acc[0][t], false, false);
      acc[1][t] = __builtin_amdgcn_wmma_f32_16x16x32_bf16(
          false, a1.v, false, bcur[t].v, (short)0, acc[1][t], false, false);
    }
#pragma unroll
    for (int t = 0; t < 4; ++t) bcur[t] = bnxt[t];
  }
  // peeled last k-step (no prefetch)
  a0.p[0] = *(const v8bf*)(a0p + kk);
  a0.p[1] = *(const v8bf*)(a0p + kk + 16);
  a1.p[0] = *(const v8bf*)(a1p + kk);
  a1.p[1] = *(const v8bf*)(a1p + kk + 16);
#pragma unroll
  for (int t = 0; t < 4; ++t) {
    acc[0][t] = __builtin_amdgcn_wmma_f32_16x16x32_bf16(
        false, a0.v, false, bcur[t].v, (short)0, acc[0][t], false, false);
    acc[1][t] = __builtin_amdgcn_wmma_f32_16x16x32_bf16(
        false, a1.v, false, bcur[t].v, (short)0, acc[1][t], false, false);
  }
}

// ---------------------------------------------------------------------------
// Fused: features -> LDS(bf16) -> WMMA GEMM1 -> lrelu -> LDS -> WMMA GEMM2
// 256 threads = 8 wave32; wave w owns rows 0..31 x cols [w*64, w*64+64).
// ---------------------------------------------------------------------------
__global__ __launch_bounds__(256) void fused_mlp(Params P) {
  __shared__ union {
    __bf16 feats[MT * FSTRIDE];
    __bf16 h[MT * HSTRIDE];
  } sm;
  __shared__ float sB1[CEDN];
  __shared__ float sB2[CEDN];

  const int tid  = threadIdx.x;
  const int lane = tid & 31;
  const int wid  = tid >> 5;          // 0..7, owns 64-col slice
  const int row0 = blockIdx.x * MT;

  for (int i = tid; i < CEDN; i += 256) { sB1[i] = P.b1[i]; sB2[i] = P.b2[i]; }

  // ---- feature tile: 32 rows x 896 cols, bf16, into LDS ----
  {
    const int r  = tid >> 3;            // 0..31  (8 threads per row)
    const int f0 = (tid & 7) * 112;     // 896/8 contiguous features each
    const int g  = row0 + r;
    const float sx = P.in[0][2*g], sy = P.in[0][2*g+1];
    const float dx = P.in[1][2*g], dy = P.in[1][2*g+1];
    const float tv  = P.in[2][g];
    const float wsv = P.in[3][g];
    const float wdv = P.in[4][g];
    __bf16* frow = &sm.feats[r * FSTRIDE];
    for (int j = 0; j < 112; ++j) {
      const int f = f0 + j;
      float val;
      if (f < 512) {
        // point encodings: [0,256)=src, [256,512)=dst; sub: 0=sin(x) 1=cos(x) 2=sin(y) 3=cos(y)
        const int base = (f < 256) ? 5 : 13;
        const int gg = f & 255;
        const int sub = gg >> 6, q = gg & 63;
        const float coord = (f < 256) ? ((sub < 2) ? sx : sy)
                                      : ((sub < 2) ? dx : dy);
        const float a = coord * P.in[base + sub*2][q] + P.in[base + sub*2 + 1][q];
        val = (sub & 1) ? __cosf(a) : __sinf(a);
      } else {
        // numeric encodings: leaky_relu(t*w + b)
        const int j2 = f - 512;
        const int which = j2 >> 7, q = j2 & 127;
        const float x = (which == 0) ? tv : (which == 1) ? wsv : wdv;
        const float a = x * P.in[21 + 2*which][q] + P.in[22 + 2*which][q];
        val = (a > 0.f) ? a : 0.01f * a;
      }
      frow[f] = (__bf16)val;
    }
  }
  __syncthreads();

  // fragment addressing (ISA 7.12.2 16-bit layouts)
  const int aRowLo = (lane & 15);             // M-tile 0 row
  const int aSel   = (lane & 16) ? 8 : 0;     // A: K chunks {0-7,16-23} / {8-15,24-31}
  const int bSel   = (lane & 16) ? 8 : 0;     // B: K chunks {0-7,16-23} / {8-15,24-31}
  const int ncol   = wid * 64 + (lane & 15);  // this wave's first N column (per lane)

  v8f acc[2][4];
#pragma unroll
  for (int m = 0; m < 2; ++m)
#pragma unroll
    for (int t = 0; t < 4; ++t) { v8f z = {}; acc[m][t] = z; }

  // ---- GEMM1: (32 x 896) @ W1 -> 32 x 512 ----
  gemm_stage<FEATN>(&sm.feats[aRowLo * FSTRIDE] + aSel,
                    &sm.feats[(16 + aRowLo) * FSTRIDE] + aSel,
                    P.W1T + (size_t)ncol * FEATN + bSel, acc);
  __syncthreads();   // all reads of feats done before h overwrites the union

  // ---- h = lrelu(acc + b1) -> LDS (bf16) ----
  {
    const int cbase = wid * 64 + (lane & 15);
#pragma unroll
    for (int m = 0; m < 2; ++m) {
      const int rbase = m * 16 + ((lane & 16) ? 8 : 0);
#pragma unroll
      for (int t = 0; t < 4; ++t) {
        const int c = cbase + t * 16;
        const float bias = sB1[c];
#pragma unroll
        for (int v = 0; v < 8; ++v) {
          float x = acc[m][t][v] + bias;
          x = (x > 0.f) ? x : 0.01f * x;
          sm.h[(rbase + v) * HSTRIDE + c] = (__bf16)x;
        }
      }
    }
  }
  __syncthreads();

  // ---- GEMM2: (32 x 512) @ W2 -> 32 x 512 ----
#pragma unroll
  for (int m = 0; m < 2; ++m)
#pragma unroll
    for (int t = 0; t < 4; ++t) { v8f z = {}; acc[m][t] = z; }

  gemm_stage<CEDN>(&sm.h[aRowLo * HSTRIDE] + aSel,
                   &sm.h[(16 + aRowLo) * HSTRIDE] + aSel,
                   P.W2T + (size_t)ncol * CEDN + bSel, acc);

  // ---- out = lrelu(acc + b2), fp32, coalesced ----
  {
    const int cbase = wid * 64 + (lane & 15);
#pragma unroll
    for (int m = 0; m < 2; ++m) {
      const int rbase = row0 + m * 16 + ((lane & 16) ? 8 : 0);
#pragma unroll
      for (int t = 0; t < 4; ++t) {
        const int c = cbase + t * 16;
        const float bias = sB2[c];
#pragma unroll
        for (int v = 0; v < 8; ++v) {
          float x = acc[m][t][v] + bias;
          x = (x > 0.f) ? x : 0.01f * x;
          P.out[(size_t)(rbase + v) * CEDN + c] = x;
        }
      }
    }
  }
}

extern "C" void kernel_launch(void* const* d_in, const int* in_sizes, int n_in,
                              void* d_out, int out_size, void* d_ws, size_t ws_size,
                              hipStream_t stream) {
  (void)in_sizes; (void)n_in; (void)out_size; (void)ws_size;

  __bf16* W1T = (__bf16*)d_ws;
  __bf16* W2T = (__bf16*)((char*)d_ws + (size_t)CEDN * FEATN * sizeof(__bf16));

  prep_weights<<<1024, 256, 0, stream>>>(
      (const float*)d_in[27], (const float*)d_in[29], W1T, W2T);

  Params P;
  for (int i = 0; i < 27; ++i) P.in[i] = (const float*)d_in[i];
  P.b1  = (const float*)d_in[28];
  P.b2  = (const float*)d_in[30];
  P.W1T = W1T;
  P.W2T = W2T;
  P.out = (float*)d_out;

  fused_mlp<<<B_ROWS / MT, 256, 0, stream>>>(P);
}
